// CrossAttentionFusion_35476429865139
// MI455X (gfx1250) — matrix operand order repaired and checked by reading, not compiled, not run on hardware
//
#include <hip/hip_runtime.h>
#include <hip/hip_bf16.h>
#include <math.h>
#include <stdint.h>

// ---------------- problem constants (match reference) ----------------
constexpr int kN = 20000;   // nodes
constexpr int kD = 256;     // embed dim
constexpr int kH = 4;       // heads
constexpr int kDH = 64;     // head dim
constexpr int kL = 2;       // layers
constexpr int kE = 320000;  // edges
constexpr int kK = 48;      // neighbor cap
constexpr float kEPS = 1e-5f;
constexpr float kNEG = -1e30f;

typedef __attribute__((ext_vector_type(16))) _Float16 v16h;
typedef __attribute__((ext_vector_type(8)))  _Float16 v8h;
typedef __attribute__((ext_vector_type(8)))  float    v8f;

// ================= neighbor table build =================
__global__ void init_nbr_kernel(int* __restrict__ nbr, int* __restrict__ pad,
                                int* __restrict__ counts) {
  int i = blockIdx.x * 256 + threadIdx.x;
  if (i < kN * kK) { nbr[i] = 0; pad[i] = 1; }
  if (i < kN) counts[i] = 0;
}

__global__ void scatter_edges_kernel(const int* __restrict__ edges,
                                     int* __restrict__ nbr, int* __restrict__ pad,
                                     int* __restrict__ counts) {
  int e = blockIdx.x * 256 + threadIdx.x;
  if (e >= kE) return;
  int s = edges[e];        // src row
  int t = edges[kE + e];   // tgt row
  int p = atomicAdd(&counts[t], 1);
  if (p < kK) { nbr[t * kK + p] = s; pad[t * kK + p] = 0; }
}

__global__ void iso_fix_kernel(const int* __restrict__ counts,
                               int* __restrict__ nbr, int* __restrict__ pad) {
  int n = blockIdx.x * 256 + threadIdx.x;
  if (n >= kN) return;
  if (counts[n] == 0) { nbr[n * kK] = n; pad[n * kK] = 0; }
}

// ================= f32 -> f16 convert =================
__global__ void cvt_f16_kernel(const float* __restrict__ in, _Float16* __restrict__ out, int n) {
  int i = blockIdx.x * 256 + threadIdx.x;
  if (i < n) out[i] = (_Float16)in[i];
}

// ================= CDNA5 async LDS copy helpers =================
// GLOBAL_LOAD_ASYNC_TO_LDS_B128 (ASYNCcnt-tracked, no VGPR staging).
// Generic shared pointers keep the LDS byte offset in addr[31:0] (ISA aperture map).
__device__ inline void async_copy_b128(void* lds, const void* gaddr) {
  unsigned l = (unsigned)(uintptr_t)lds;
  unsigned long long g = (unsigned long long)(uintptr_t)gaddr;
  asm volatile("global_load_async_to_lds_b128 %0, %1, off"
               :: "v"(l), "v"(g) : "memory");
}
__device__ inline void wait_async_le4() {
  asm volatile("s_wait_asynccnt 0x4" ::: "memory");
}
__device__ inline void wait_async_0() {
  asm volatile("s_wait_asynccnt 0x0" ::: "memory");
}

// ================= WMMA GEMM =================
// C[M,N] = A16[M,K] * W16[N,K]^T + bias[N]   (W stored row-major: row j holds K values)
// EPI==0: f32 out.  EPI==1: exact GELU, f16 out.
__device__ inline v16h frag_combine(v8h lo, v8h hi) {
  v16h r;
#pragma unroll
  for (int i = 0; i < 8; ++i) { r[i] = lo[i]; r[i + 8] = hi[i]; }
  return r;
}

// A fragment per ISA 16-bit A(16x32) table: lanes0-15 K={0..7,16..23}, lanes16-31 K={8..15,24..31}
__device__ inline v16h load_fragA(const _Float16* lsA, int lane, int m0) {
  const _Float16* p = lsA + (m0 + (lane & 15)) * 32 + ((lane >> 4) << 3);
  v8h lo = *(const v8h*)p;
  v8h hi = *(const v8h*)(p + 16);
  return frag_combine(lo, hi);
}

// B fragment: lanes0-15 K={0..7,16..23}, lanes16-31 K={8..15,24..31} (col = lane&15)
__device__ inline v16h load_fragB(const _Float16* lsB, int lane, int n0) {
  const _Float16* p = lsB + (n0 + (lane & 15)) * 32 + ((lane >> 4) << 3);
  v8h lo = *(const v8h*)p;
  v8h hi = *(const v8h*)(p + 16);
  return frag_combine(lo, hi);
}

template <int EPI>
__global__ __launch_bounds__(128) void gemm_wmma_kernel(
    const _Float16* __restrict__ A, const _Float16* __restrict__ W,
    const float* __restrict__ bias, void* __restrict__ out,
    int Mrows, int Ncols, int Kdim) {
  __shared__ _Float16 lsA[2][64 * 32];
  __shared__ _Float16 lsB[2][64 * 32];

  const int tid  = threadIdx.x;
  const int lane = tid & 31;
  const int wid  = tid >> 5;        // 4 waves: each owns a 16-row band
  const int bm   = blockIdx.y * 64;
  const int bn   = blockIdx.x * 64;

  const int srow = tid >> 1;          // 0..63 staging row
  const int soff = (tid & 1) * 16;    // 0 or 16 halves

  v8f acc[4];
#pragma unroll
  for (int s = 0; s < 4; ++s)
#pragma unroll
    for (int i = 0; i < 8; ++i) acc[s][i] = 0.0f;

  int gRow = bm + srow; if (gRow >= Mrows) gRow = Mrows - 1;
  const int gCol = bn + srow;
  const _Float16* gA = A + (size_t)gRow * Kdim + soff;
  const _Float16* gB = W + (size_t)gCol * Kdim + soff;

  // stage one 64x32 A tile + 64x32 B tile into LDS buffer `buf` (4 async b128/thread)
  auto stage = [&](int buf, int kk) {
    async_copy_b128(&lsA[buf][srow * 32 + soff],     gA + kk);
    async_copy_b128(&lsA[buf][srow * 32 + soff + 8], gA + kk + 8);
    async_copy_b128(&lsB[buf][srow * 32 + soff],     gB + kk);
    async_copy_b128(&lsB[buf][srow * 32 + soff + 8], gB + kk + 8);
  };
  auto compute = [&](int buf) {
    v16h af = load_fragA(&lsA[buf][0], lane, wid * 16);
#pragma unroll
    for (int s = 0; s < 4; ++s) {
      v16h bf = load_fragB(&lsB[buf][0], lane, s * 16);
      acc[s] = __builtin_amdgcn_wmma_f32_16x16x32_f16(
          false, af, false, bf, (short)0, acc[s], false, false);
    }
  };

  const int steps = Kdim >> 5;
  stage(0, 0);
  for (int k = 0; k < steps - 1; ++k) {
    stage((k + 1) & 1, (k + 1) * 32);   // prefetch next tile (async)
    wait_async_le4();                   // in-order: tile k's 4 copies are done
    __syncthreads();                    // ...for every wave in the block
    compute(k & 1);
    __syncthreads();                    // done reading buf before it is rewritten
  }
  wait_async_0();
  __syncthreads();
  compute((steps - 1) & 1);

  // epilogue: VGPR r of C holds M = r + (lane>=16 ? 8 : 0), N = lane&15
  const int rbase = bm + wid * 16 + ((lane >> 4) << 3);
  const int cl = lane & 15;
#pragma unroll
  for (int s = 0; s < 4; ++s) {
    const int col = bn + s * 16 + cl;
    const float bv = bias[col];
#pragma unroll
    for (int r = 0; r < 8; ++r) {
      const int row = rbase + r;
      if (row < Mrows) {
        float v = acc[s][r] + bv;
        if (EPI == 1) {
          v = 0.5f * v * (1.0f + erff(v * 0.70710678118654752f));
          ((_Float16*)out)[(size_t)row * Ncols + col] = (_Float16)v;
        } else {
          ((float*)out)[(size_t)row * Ncols + col] = v;
        }
      }
    }
  }
}

// ================= per-node neighbor attention =================
__global__ __launch_bounds__(256) void attn_kernel(
    const float* __restrict__ q, const float* __restrict__ kv,
    const int* __restrict__ nbr, const int* __restrict__ pad,
    _Float16* __restrict__ ao16) {
  __shared__ float sQ[kD];
  __shared__ float sS[kH * kK];
  __shared__ int   sN[kK];
  __shared__ int   sP[kK];

  const int n = blockIdx.x;
  const int tid = threadIdx.x;

  sQ[tid] = q[(size_t)n * kD + tid];
  if (tid < kK) { sN[tid] = nbr[n * kK + tid]; sP[tid] = pad[n * kK + tid]; }
  __syncthreads();

  if (tid < kH * kK) {                      // 192 score dots
    const int h = tid / kK, k = tid % kK;
    float s = kNEG;
    if (!sP[k]) {
      const float* kr = kv + (size_t)sN[k] * (2 * kD) + h * kDH;
      const float* qr = sQ + h * kDH;
      float acc = 0.0f;
#pragma unroll 8
      for (int d = 0; d < kDH; ++d) acc += qr[d] * kr[d];
      s = acc * 0.125f;                     // 1/sqrt(64)
    }
    sS[h * kK + k] = s;
  }
  __syncthreads();

  if (tid < kH) {                           // softmax per head over 48 slots
    float m = kNEG;
    for (int k = 0; k < kK; ++k) m = fmaxf(m, sS[tid * kK + k]);
    float sum = 0.0f;
    for (int k = 0; k < kK; ++k) {
      float e = __expf(sS[tid * kK + k] - m);
      sS[tid * kK + k] = e; sum += e;
    }
    float inv = 1.0f / sum;
    for (int k = 0; k < kK; ++k) sS[tid * kK + k] *= inv;
  }
  __syncthreads();

  const int h = tid >> 6;                   // output dim tid -> head
  float acc = 0.0f;
  for (int k = 0; k < kK; ++k) {
    const float w = sS[h * kK + k];
    acc += w * kv[(size_t)sN[k] * (2 * kD) + kD + tid];   // V half of kv
  }
  ao16[(size_t)n * kD + tid] = (_Float16)acc;
}

// ================= fused residual + LayerNorm (+ f16 copy-out) =================
__global__ __launch_bounds__(256) void residual_ln_kernel(
    const float* __restrict__ x, const float* __restrict__ t,
    const float* __restrict__ g, const float* __restrict__ b,
    float* __restrict__ xout, _Float16* __restrict__ x16out) {
  __shared__ float red[kD];
  const int n = blockIdx.x, tid = threadIdx.x;
  const size_t idx = (size_t)n * kD + tid;

  float v = x[idx] + t[idx];
  red[tid] = v;
  __syncthreads();
#pragma unroll
  for (int s = 128; s > 0; s >>= 1) { if (tid < s) red[tid] += red[tid + s]; __syncthreads(); }
  const float mean = red[0] * (1.0f / kD);
  __syncthreads();
  const float dv = v - mean;
  red[tid] = dv * dv;
  __syncthreads();
#pragma unroll
  for (int s = 128; s > 0; s >>= 1) { if (tid < s) red[tid] += red[tid + s]; __syncthreads(); }
  const float var = red[0] * (1.0f / kD);
  const float y = dv * rsqrtf(var + kEPS) * g[tid] + b[tid];
  xout[idx] = y;
  x16out[idx] = (_Float16)y;
}

// ================= host orchestration =================
extern "C" void kernel_launch(void* const* d_in, const int* in_sizes, int n_in,
                              void* d_out, int out_size, void* d_ws, size_t ws_size,
                              hipStream_t stream) {
  const float* expr      = (const float*)d_in[0];
  const float* spat      = (const float*)d_in[1];
  const float* in_proj_w = (const float*)d_in[2];   // (L,768,256)
  const float* in_proj_b = (const float*)d_in[3];   // (L,768)
  const float* out_proj_w= (const float*)d_in[4];   // (L,256,256)
  const float* out_proj_b= (const float*)d_in[5];   // (L,256)
  const float* ffn_w1    = (const float*)d_in[6];   // (L,512,256)
  const float* ffn_b1    = (const float*)d_in[7];   // (L,512)
  const float* ffn_w2    = (const float*)d_in[8];   // (L,256,512)
  const float* ffn_b2    = (const float*)d_in[9];   // (L,256)
  const float* ln1_g = (const float*)d_in[10];
  const float* ln1_b = (const float*)d_in[11];
  const float* ln2_g = (const float*)d_in[12];
  const float* ln2_b = (const float*)d_in[13];
  const int*   edges = (const int*)d_in[14];        // (2,E)

  // carve workspace
  size_t off = 0;
  auto carve = [&](size_t bytes) -> void* {
    void* p = (char*)d_ws + off;
    off += (bytes + 255) & ~(size_t)255;
    return p;
  };
  float*    x     = (float*)   carve((size_t)kN * kD * 4);
  _Float16* x16   = (_Float16*)carve((size_t)kN * kD * 2);
  _Float16* sp16  = (_Float16*)carve((size_t)kN * kD * 2);
  float*    q     = (float*)   carve((size_t)kN * kD * 4);
  float*    kv    = (float*)   carve((size_t)kN * 2 * kD * 4);
  _Float16* ao16  = (_Float16*)carve((size_t)kN * kD * 2);
  float*    tmp   = (float*)   carve((size_t)kN * kD * 4);
  _Float16* h16   = (_Float16*)carve((size_t)kN * 2 * kD * 2);
  _Float16* w16_in  = (_Float16*)carve((size_t)kL * 3 * kD * kD * 2);
  _Float16* w16_out = (_Float16*)carve((size_t)kL * kD * kD * 2);
  _Float16* w16_f1  = (_Float16*)carve((size_t)kL * 2 * kD * kD * 2);
  _Float16* w16_f2  = (_Float16*)carve((size_t)kL * kD * 2 * kD * 2);
  int* nbr    = (int*)carve((size_t)kN * kK * 4);
  int* pad    = (int*)carve((size_t)kN * kK * 4);
  int* counts = (int*)carve((size_t)kN * 4);

  auto blk = [](int n) { return (n + 255) / 256; };

  // 1) neighbor table
  init_nbr_kernel<<<blk(kN * kK), 256, 0, stream>>>(nbr, pad, counts);
  scatter_edges_kernel<<<blk(kE), 256, 0, stream>>>(edges, nbr, pad, counts);
  iso_fix_kernel<<<blk(kN), 256, 0, stream>>>(counts, nbr, pad);

  // 2) f16 conversions (weights + embeds), x init
  cvt_f16_kernel<<<blk(kN * kD), 256, 0, stream>>>(spat, sp16, kN * kD);
  cvt_f16_kernel<<<blk(kN * kD), 256, 0, stream>>>(expr, x16, kN * kD);
  hipMemcpyAsync(x, expr, (size_t)kN * kD * 4, hipMemcpyDeviceToDevice, stream);
  cvt_f16_kernel<<<blk(kL * 3 * kD * kD), 256, 0, stream>>>(in_proj_w,  w16_in,  kL * 3 * kD * kD);
  cvt_f16_kernel<<<blk(kL * kD * kD),     256, 0, stream>>>(out_proj_w, w16_out, kL * kD * kD);
  cvt_f16_kernel<<<blk(kL * 2 * kD * kD), 256, 0, stream>>>(ffn_w1,     w16_f1,  kL * 2 * kD * kD);
  cvt_f16_kernel<<<blk(kL * kD * 2 * kD), 256, 0, stream>>>(ffn_w2,     w16_f2,  kL * kD * 2 * kD);

  const int MT = (kN + 63) / 64;  // 313 M-tiles

  for (int l = 0; l < kL; ++l) {
    const _Float16* Wl_in = w16_in + (size_t)l * 3 * kD * kD;
    // q = x @ Wq^T + bq
    gemm_wmma_kernel<0><<<dim3(kD / 64, MT), 128, 0, stream>>>(
        x16, Wl_in, in_proj_b + (size_t)l * 3 * kD, q, kN, kD, kD);
    // kv = spatial @ [Wk;Wv]^T + [bk;bv]   (rows 256..767 of in_proj)
    gemm_wmma_kernel<0><<<dim3(2 * kD / 64, MT), 128, 0, stream>>>(
        sp16, Wl_in + (size_t)kD * kD, in_proj_b + (size_t)l * 3 * kD + kD,
        kv, kN, 2 * kD, kD);
    // neighbor attention -> ao (f16)
    attn_kernel<<<kN, 256, 0, stream>>>(q, kv, nbr, pad, ao16);
    // out_proj
    gemm_wmma_kernel<0><<<dim3(kD / 64, MT), 128, 0, stream>>>(
        ao16, w16_out + (size_t)l * kD * kD, out_proj_b + (size_t)l * kD,
        tmp, kN, kD, kD);
    // x = LN1(x + attn_out)
    residual_ln_kernel<<<kN, 256, 0, stream>>>(
        x, tmp, ln1_g + (size_t)l * kD, ln1_b + (size_t)l * kD, x, x16);
    // h = gelu(x @ W1^T + b1)  (f16 out)
    gemm_wmma_kernel<1><<<dim3(2 * kD / 64, MT), 128, 0, stream>>>(
        x16, w16_f1 + (size_t)l * 2 * kD * kD, ffn_b1 + (size_t)l * 2 * kD,
        h16, kN, 2 * kD, kD);
    // tmp = h @ W2^T + b2
    gemm_wmma_kernel<0><<<dim3(kD / 64, MT), 128, 0, stream>>>(
        h16, w16_f2 + (size_t)l * kD * 2 * kD, ffn_b2 + (size_t)l * kD,
        tmp, kN, kD, 2 * kD);
    // x = LN2(x + tmp); final layer writes straight to d_out
    float* xdst = (l == kL - 1) ? (float*)d_out : x;
    residual_ln_kernel<<<kN, 256, 0, stream>>>(
        x, tmp, ln2_g + (size_t)l * kD, ln2_b + (size_t)l * kD, xdst, x16);
  }
}
